// TemporalGATEncoder_28853590295054
// MI455X (gfx1250) — compile-verified, matrix operand order checked by reading, not run on hardware
//
#include <hip/hip_runtime.h>
#include <hip/hip_bf16.h>
#include <math.h>

// ---------------------------------------------------------------------------
// Types for CDNA5 WMMA
// ---------------------------------------------------------------------------
typedef __attribute__((ext_vector_type(16))) _Float16 v16h;
typedef __attribute__((ext_vector_type(8)))  _Float16 v8h;
typedef __attribute__((ext_vector_type(8)))  float    v8f;

union V16U { v16h v; v8h h[2]; };

// Async global->LDS 16B copy (per active lane), CDNA5 (CDNA5 ISA).
// VDST = 32-bit LDS byte offset, VADDR = 64-bit global address.
__device__ __forceinline__ void async_g2l_b128(unsigned lds_off, const void* gaddr) {
    asm volatile("global_load_async_to_lds_b128 %0, %1, off"
                 :: "v"(lds_off), "v"(gaddr) : "memory");
}
// Low 32 bits of a generic pointer into LDS == LDS byte offset (aperture in high dword).
__device__ __forceinline__ unsigned lds_off32(const void* p) {
    return (unsigned)(unsigned long long)p;
}

// ---------------------------------------------------------------------------
// f16-in / f32-out GEMM with f16 WMMA core.
//   C[M x Nc] = A[M x K] * Bt^T (+bias), A row-major, Bt is Nc x K
//   -> BOTH tiles are K-contiguous -> async b128 global->LDS tile loads.
// Block tile 128x64, 8 waves (wave32), each wave a 32x32 tile via
// 2x2 v_wmma_f32_16x16x32_f16. Double-buffered LDS + ASYNCcnt pipelining:
// each tile is exactly 3 async instrs/wave; async loads complete in order,
// so s_wait_asynccnt 3 releases tile kb while tile kb+1 streams.
// ---------------------------------------------------------------------------
#define BM 128
#define BN 64
#define BK 32
#define LDS_AS 40   // halves per row (80B, multiple of 16B -> aligned b128)
#define LDS_BS 40

__global__ __launch_bounds__(256)
void k_gemm_wmma(const _Float16* __restrict__ A, int lda,
                 const _Float16* __restrict__ Bt, int ldb,
                 float* __restrict__ C, int ldc,
                 int M, int Nc, int K, const float* __restrict__ bias)
{
    __shared__ __align__(16) _Float16 As[2][BM * LDS_AS];
    __shared__ __align__(16) _Float16 Bs[2][BN * LDS_BS];

    const int tid   = threadIdx.x;
    const int lane  = tid & 31;        // wave32
    const int wave  = tid >> 5;        // 0..7
    const int wm    = wave & 3;        // 4 waves along M
    const int wn    = wave >> 2;       // 2 waves along N
    const int lrow  = lane & 15;
    const int lhalf = lane >> 4;

    const int rowBase = blockIdx.y * BM;
    const int colBase = blockIdx.x * BN;
    const bool tileFull = (rowBase + BM <= M) && (colBase + BN <= Nc);
    const bool allFast  = tileFull && (K % BK == 0);

    v8f zero = {};
    v8f acc[2][2];
    acc[0][0] = zero; acc[0][1] = zero; acc[1][0] = zero; acc[1][1] = zero;

    const int nkb = (K + BK - 1) / BK;

    // Fragment load + 2x2 WMMA from a given LDS buffer.
    auto computeTile = [&](const _Float16* Asb, const _Float16* Bsb) {
        // ISA 16-bit A/B layout: lane 0-15 -> row/col = lane,
        // VGPR0-3 = halves [half*8 .. +7], VGPR4-7 = [16+half*8 .. +7]
        // -> two aligned 16B LDS reads each.
        V16U aF[2], bF[2];
        for (int mi = 0; mi < 2; ++mi) {
            int r = wm * 32 + mi * 16 + lrow;
            const _Float16* p = &Asb[r * LDS_AS + lhalf * 8];
            aF[mi].h[0] = *(const v8h*)(p);
            aF[mi].h[1] = *(const v8h*)(p + 16);
        }
        for (int ni = 0; ni < 2; ++ni) {
            int c = wn * 32 + ni * 16 + lrow;
            const _Float16* p = &Bsb[c * LDS_BS + lhalf * 8];
            bF[ni].h[0] = *(const v8h*)(p);
            bF[ni].h[1] = *(const v8h*)(p + 16);
        }
        for (int mi = 0; mi < 2; ++mi)
            for (int ni = 0; ni < 2; ++ni)
                acc[mi][ni] = __builtin_amdgcn_wmma_f32_16x16x32_f16(
                    false, aF[mi].v, false, bF[ni].v,
                    (short)0, acc[mi][ni], false, false);
    };

    if (allFast) {
        // Issue one tile's async copies: 2 A-instrs + 1 B-instr per thread/wave.
        auto issueTile = [&](int kb, int buf) {
            const int k0 = kb * BK;
            const unsigned aB = lds_off32(&As[buf][0]);
            const unsigned bB = lds_off32(&Bs[buf][0]);
            // A tile: 128 rows x 4 chunks of 16B (K-contiguous in memory).
            for (int c = tid; c < BM * 4; c += 256) {
                int r = c >> 2, kq = c & 3;
                unsigned l = aB + (unsigned)(r * (LDS_AS * 2) + kq * 16);
                const void* g = (const void*)(A + (size_t)(rowBase + r) * lda + k0 + kq * 8);
                async_g2l_b128(l, g);
            }
            // B tile: 64 rows x 4 chunks of 16B.
            for (int c = tid; c < BN * 4; c += 256) {
                int r = c >> 2, kq = c & 3;
                unsigned l = bB + (unsigned)(r * (LDS_BS * 2) + kq * 16);
                const void* g = (const void*)(Bt + (size_t)(colBase + r) * ldb + k0 + kq * 8);
                async_g2l_b128(l, g);
            }
        };

        issueTile(0, 0);
        for (int kb = 0; kb < nkb; ++kb) {
            const int cur = kb & 1;
            if (kb + 1 < nkb) {
                issueTile(kb + 1, cur ^ 1);   // prefetch next tile into other buffer
                // In-order async completion: <=3 outstanding == tile kb done.
                asm volatile("s_wait_asynccnt 3" ::: "memory");
            } else {
                asm volatile("s_wait_asynccnt 0" ::: "memory");
            }
            __syncthreads();                  // tile kb visible to all waves
            computeTile(As[cur], Bs[cur]);
            __syncthreads();                  // all reads done before buffer reuse
        }
    } else {
        // Generic guarded path (K tails / ragged tiles), single buffer.
        for (int kb = 0; kb < nkb; ++kb) {
            const int k0 = kb * BK;
            __syncthreads();
            for (int i = tid; i < BM * BK; i += 256) {
                int r = i >> 5, k = i & 31;
                int gr = rowBase + r, gk = k0 + k;
                As[0][r * LDS_AS + k] = (gr < M && gk < K) ? A[(size_t)gr * lda + gk]
                                                           : (_Float16)0.0f;
            }
            for (int i = tid; i < BN * BK; i += 256) {
                int n = i >> 5, k = i & 31;
                int gn = colBase + n, gk = k0 + k;
                Bs[0][n * LDS_BS + k] = (gn < Nc && gk < K) ? Bt[(size_t)gn * ldb + gk]
                                                            : (_Float16)0.0f;
            }
            __syncthreads();
            computeTile(As[0], Bs[0]);
        }
    }

    // Epilogue: C/D layout -> lane 0-15: N=lane, M=vgpr(+8 for lanes 16-31).
    for (int mi = 0; mi < 2; ++mi)
        for (int ni = 0; ni < 2; ++ni) {
            int rb = rowBase + wm * 32 + mi * 16 + lhalf * 8;
            int cb = colBase + wn * 32 + ni * 16 + lrow;
            if (cb < Nc) {
                float badd = bias ? bias[cb] : 0.0f;
                for (int i = 0; i < 8; ++i) {
                    int rr = rb + i;
                    if (rr < M) C[(size_t)rr * ldc + cb] = acc[mi][ni][i] + badd;
                }
            }
        }
}

// ---------------------------------------------------------------------------
// f32 -> f16 staging (straight and transposing variants)
// ---------------------------------------------------------------------------
__global__ void k_cvt_f16(const float* __restrict__ src, _Float16* __restrict__ dst, size_t n) {
    size_t i = (size_t)blockIdx.x * blockDim.x + threadIdx.x;
    if (i < n) dst[i] = (_Float16)src[i];
}
__global__ void k_cvt_f16_T(const float* __restrict__ src, _Float16* __restrict__ dst,
                            int R, int Ccols) {   // src R x C -> dst C x R
    size_t i = (size_t)blockIdx.x * blockDim.x + threadIdx.x;
    if (i >= (size_t)R * Ccols) return;
    int r = (int)(i / Ccols), c = (int)(i - (size_t)r * Ccols);
    dst[(size_t)c * R + r] = (_Float16)src[i];
}

// ---------------------------------------------------------------------------
// Order-preserving float <-> uint key for atomicMax on floats
// ---------------------------------------------------------------------------
__device__ __forceinline__ unsigned fkey(float f) {
    unsigned u = __float_as_uint(f);
    return (u & 0x80000000u) ? ~u : (u | 0x80000000u);
}
__device__ __forceinline__ float fdec(unsigned k) {
    unsigned u = (k & 0x80000000u) ? (k & 0x7fffffffu) : ~k;
    return __uint_as_float(u);
}

// ---------------------------------------------------------------------------
// Fills
// ---------------------------------------------------------------------------
__global__ void k_fill_f32(float* p, float v, size_t n) {
    size_t i = (size_t)blockIdx.x * blockDim.x + threadIdx.x;
    if (i < n) p[i] = v;
}
__global__ void k_fill_u32(unsigned* p, unsigned v, size_t n) {
    size_t i = (size_t)blockIdx.x * blockDim.x + threadIdx.x;
    if (i < n) p[i] = v;
}
__global__ void k_fill_bias(float* out, const float* __restrict__ bias, size_t total, int C) {
    size_t i = (size_t)blockIdx.x * blockDim.x + threadIdx.x;
    if (i < total) out[i] = bias[i % C];
}

// ---------------------------------------------------------------------------
// Per-node attention scores (one wave per (node, head), shfl-xor reduce)
// ---------------------------------------------------------------------------
__global__ __launch_bounds__(256)
void k_node_scores(const float* __restrict__ feat, const float* __restrict__ a_s,
                   const float* __restrict__ a_d, float* __restrict__ ssrc,
                   float* __restrict__ sdst, int Nn, int H, int F)
{
    int wid  = blockIdx.x * (blockDim.x >> 5) + (threadIdx.x >> 5);
    int lane = threadIdx.x & 31;
    if (wid >= Nn * H) return;
    int n = wid / H, h = wid - n * H;
    const float* f  = feat + (size_t)n * H * F + (size_t)h * F;
    const float* as = a_s + h * F;
    const float* ad = a_d + h * F;
    float ss = 0.f, sd = 0.f;
    for (int j = lane; j < F; j += 32) {
        float v = f[j];
        ss += v * as[j];
        sd += v * ad[j];
    }
    for (int off = 16; off > 0; off >>= 1) {
        ss += __shfl_xor(ss, off);
        sd += __shfl_xor(sd, off);
    }
    if (lane == 0) { ssrc[wid] = ss; sdst[wid] = sd; }
}

// ---------------------------------------------------------------------------
// Edge pass 1: per-dst-per-head running max of leaky_relu(score)
// ---------------------------------------------------------------------------
__global__ void k_edge_max(const float* __restrict__ ssrc, const float* __restrict__ sdst,
                           const int* __restrict__ ei, int E, int ETOT, int H,
                           unsigned* __restrict__ mkey)
{
    int e = blockIdx.x * blockDim.x + threadIdx.x;
    if (e >= ETOT) return;
    int s, d;
    if (e < E) { s = ei[e]; d = ei[E + e]; } else { s = d = e - E; }
    for (int h = 0; h < H; ++h) {
        float el = ssrc[s * H + h] + sdst[d * H + h];
        el = el > 0.f ? el : 0.2f * el;
        atomicMax(&mkey[d * H + h], fkey(el));
    }
}

// ---------------------------------------------------------------------------
// Edge pass 2: alpha = exp(el - max[dst]); per-dst sums
// ---------------------------------------------------------------------------
__global__ void k_edge_expsum(const float* __restrict__ ssrc, const float* __restrict__ sdst,
                              const int* __restrict__ ei, int E, int ETOT, int H,
                              const unsigned* __restrict__ mkey, float* __restrict__ ssum,
                              float* __restrict__ alpha)
{
    int e = blockIdx.x * blockDim.x + threadIdx.x;
    if (e >= ETOT) return;
    int s, d;
    if (e < E) { s = ei[e]; d = ei[E + e]; } else { s = d = e - E; }
    for (int h = 0; h < H; ++h) {
        float el = ssrc[s * H + h] + sdst[d * H + h];
        el = el > 0.f ? el : 0.2f * el;
        float a = expf(el - fdec(mkey[d * H + h]));
        alpha[(size_t)e * H + h] = a;
        atomicAdd(&ssum[d * H + h], a);
    }
}

// ---------------------------------------------------------------------------
// Edge pass 3: out[dst] += feat[src] * alpha/sum; one wave per edge,
// 32 lanes stride over H*F features (atomics resolve in L2: out fits in 192MB)
// ---------------------------------------------------------------------------
__global__ __launch_bounds__(256)
void k_edge_aggregate(const float* __restrict__ feat, const float* __restrict__ alpha,
                      const float* __restrict__ ssum, const int* __restrict__ ei,
                      int E, int ETOT, int H, int F, float* __restrict__ out)
{
    int wid  = blockIdx.x * (blockDim.x >> 5) + (threadIdx.x >> 5);
    int lane = threadIdx.x & 31;
    if (wid >= ETOT) return;
    int s, d;
    if (wid < E) { s = ei[wid]; d = ei[E + wid]; } else { s = d = wid - E; }
    const int HF = H * F;
    float coef[4];
    for (int h = 0; h < H; ++h)
        coef[h] = alpha[(size_t)wid * H + h] / (ssum[d * H + h] + 1e-16f);
    const float* fs = feat + (size_t)s * HF;
    float*       od = out  + (size_t)d * HF;
    for (int j = lane; j < HF; j += 32)
        atomicAdd(&od[j], fs[j] * coef[j / F]);
}

// ---------------------------------------------------------------------------
// BatchNorm (training stats) + ELU
// ---------------------------------------------------------------------------
__global__ void k_bn_partial(const float* __restrict__ X, int Nrows, int C,
                             float* __restrict__ gsum, float* __restrict__ gsq, int rpb)
{
    int c  = threadIdx.x;          // blockDim.x == C
    int r0 = blockIdx.x * rpb;
    int r1 = r0 + rpb; if (r1 > Nrows) r1 = Nrows;
    float s = 0.f, q = 0.f;
    for (int r = r0; r < r1; ++r) {
        float v = X[(size_t)r * C + c];
        s += v; q += v * v;
    }
    atomicAdd(&gsum[c], s);
    atomicAdd(&gsq[c], q);
}

__global__ void k_bn_stats(float* gsum, float* gsq, int Nrows, int C)
{
    int c = threadIdx.x;
    if (c >= C) return;
    float mean = gsum[c] / (float)Nrows;
    float var  = gsq[c] / (float)Nrows - mean * mean;
    gsum[c] = mean;
    gsq[c]  = rsqrtf(var + 1e-5f);
}

__global__ void k_bn_elu(float* __restrict__ X, const float* __restrict__ meanA,
                         const float* __restrict__ rstdA, const float* __restrict__ g,
                         const float* __restrict__ be, size_t total, int C)
{
    size_t i = (size_t)blockIdx.x * blockDim.x + threadIdx.x;
    if (i >= total) return;
    int c = (int)(i % C);
    float y = (X[i] - meanA[c]) * rstdA[c] * g[c] + be[c];
    X[i] = y > 0.f ? y : (expf(y) - 1.f);
}

// ---------------------------------------------------------------------------
// GRU gate update (torch gate math); gi precomputed for all t
// ---------------------------------------------------------------------------
__global__ void k_gru_gates(const float* __restrict__ gi, const float* __restrict__ gh,
                            const float* __restrict__ bih, const float* __restrict__ bhh,
                            float* __restrict__ hstate, int BC, int Cch, int t, int Tt)
{
    int i = blockIdx.x * blockDim.x + threadIdx.x;
    if (i >= BC * 128) return;
    int s = i >> 7, j = i & 127;
    int b = s / Cch, c = s - b * Cch;
    int row = (b * Tt + t) * Cch + c;          // node index for (b, t, c)
    float ir  = gi[(size_t)row * 384 + j]       + bih[j];
    float iz  = gi[(size_t)row * 384 + 128 + j] + bih[128 + j];
    float in_ = gi[(size_t)row * 384 + 256 + j] + bih[256 + j];
    float hr  = gh[(size_t)s * 384 + j]         + bhh[j];
    float hz  = gh[(size_t)s * 384 + 128 + j]   + bhh[128 + j];
    float hn  = gh[(size_t)s * 384 + 256 + j]   + bhh[256 + j];
    float r = 1.f / (1.f + expf(-(ir + hr)));
    float z = 1.f / (1.f + expf(-(iz + hz)));
    float nt = tanhf(in_ + r * hn);
    float hp = hstate[(size_t)s * 128 + j];
    hstate[(size_t)s * 128 + j] = (1.f - z) * nt + z * hp;
}

// ---------------------------------------------------------------------------
// Channel-attention pooling + final projection
// ---------------------------------------------------------------------------
__global__ __launch_bounds__(64)
void k_pool_score(const float* __restrict__ hst, const float* __restrict__ Wp1,
                  const float* __restrict__ bp1, const float* __restrict__ Wp2,
                  const float* __restrict__ bp2, float* __restrict__ score)
{
    __shared__ float hrow[128];
    __shared__ float red[64];
    int s = blockIdx.x, j = threadIdx.x;
    hrow[j]      = hst[(size_t)s * 128 + j];
    hrow[j + 64] = hst[(size_t)s * 128 + 64 + j];
    __syncthreads();
    float a = 0.f;
    for (int k = 0; k < 128; ++k) a += hrow[k] * Wp1[k * 64 + j];
    red[j] = tanhf(a + bp1[j]) * Wp2[j];
    __syncthreads();
    for (int off = 32; off > 0; off >>= 1) {
        if (j < off) red[j] += red[j + off];
        __syncthreads();
    }
    if (j == 0) score[s] = red[0] + bp2[0];
}

__global__ __launch_bounds__(128)
void k_pool_apply(const float* __restrict__ score, const float* __restrict__ hst,
                  int Cch, float* __restrict__ pooled)
{
    __shared__ float w[96];
    int b = blockIdx.x, f = threadIdx.x;
    if (f < Cch) w[f] = score[b * Cch + f];
    __syncthreads();
    if (f == 0) {
        float m = -1e30f;
        for (int c = 0; c < Cch; ++c) m = w[c] > m ? w[c] : m;
        float sum = 0.f;
        for (int c = 0; c < Cch; ++c) { float e = expf(w[c] - m); w[c] = e; sum += e; }
        float inv = 1.f / sum;
        for (int c = 0; c < Cch; ++c) w[c] *= inv;
    }
    __syncthreads();
    float acc = 0.f;
    for (int c = 0; c < Cch; ++c)
        acc += w[c] * hst[((size_t)(b * Cch + c)) * 128 + f];
    pooled[(size_t)b * 128 + f] = acc;
}

__global__ __launch_bounds__(64)
void k_final(const float* __restrict__ pooled, const float* __restrict__ We,
             const float* __restrict__ bemb, float* __restrict__ out)
{
    __shared__ float p[128];
    int b = blockIdx.x, e = threadIdx.x;
    p[e]      = pooled[(size_t)b * 128 + e];
    p[e + 64] = pooled[(size_t)b * 128 + 64 + e];
    __syncthreads();
    float acc = bemb[e];
    for (int k = 0; k < 128; ++k) acc += p[k] * We[k * 64 + e];
    out[(size_t)b * 64 + e] = acc;
}

// ---------------------------------------------------------------------------
// Host launcher
// ---------------------------------------------------------------------------
extern "C" void kernel_launch(void* const* d_in, const int* in_sizes, int n_in,
                              void* d_out, int out_size, void* d_ws, size_t ws_size,
                              hipStream_t stream)
{
    (void)n_in; (void)out_size; (void)ws_size;

    const float* x    = (const float*)d_in[0];
    const int*   ei   = (const int*)d_in[1];
    const float* W1   = (const float*)d_in[2];
    const float* as1  = (const float*)d_in[3];
    const float* ad1  = (const float*)d_in[4];
    const float* b1   = (const float*)d_in[5];
    const float* W2   = (const float*)d_in[6];
    const float* as2  = (const float*)d_in[7];
    const float* ad2  = (const float*)d_in[8];
    const float* b2   = (const float*)d_in[9];
    const float* W3   = (const float*)d_in[10];
    const float* as3  = (const float*)d_in[11];
    const float* ad3  = (const float*)d_in[12];
    const float* b3   = (const float*)d_in[13];
    const float* g1   = (const float*)d_in[14];
    const float* be1  = (const float*)d_in[15];
    const float* g2   = (const float*)d_in[16];
    const float* be2  = (const float*)d_in[17];
    const float* g3   = (const float*)d_in[18];
    const float* be3  = (const float*)d_in[19];
    const float* Wih  = (const float*)d_in[20];
    const float* Whh  = (const float*)d_in[21];
    const float* bih  = (const float*)d_in[22];
    const float* bhh  = (const float*)d_in[23];
    const float* Wp1  = (const float*)d_in[24];
    const float* bp1  = (const float*)d_in[25];
    const float* Wp2  = (const float*)d_in[26];
    const float* bp2  = (const float*)d_in[27];
    const float* We   = (const float*)d_in[28];
    const float* bemb = (const float*)d_in[29];

    const int INF  = 5;
    const int Nn   = in_sizes[0] / INF;      // 63744 (= 498 * 128)
    const int E    = in_sizes[1] / 2;        // 512000
    const int ETOT = E + Nn;
    const int Bb = 256, Tt = 3, Cch = 83;
    const int BC = Bb * Cch;                 // 21248 (= 166 * 128)
    const int HF = 512;

    // --- carve workspace ---
    char* w = (char*)d_ws;
    auto carve = [&](size_t bytes) -> void* {
        void* p = (void*)w;
        w += (bytes + 255) & ~(size_t)255;
        return p;
    };
    float*     P      = (float*)carve((size_t)Nn * 512 * 4);   // lin features / gi_all
    float*     Q      = (float*)carve((size_t)Nn * 512 * 4);   // aggregated / h
    float*     ssrc   = (float*)carve((size_t)Nn * 4 * 4);
    float*     sdst   = (float*)carve((size_t)Nn * 4 * 4);
    unsigned*  mkey   = (unsigned*)carve((size_t)Nn * 4 * 4);
    float*     ssum   = (float*)carve((size_t)Nn * 4 * 4);
    float*     alphaW = (float*)carve((size_t)ETOT * 4 * 4);
    float*     gsum   = (float*)carve(512 * 4);
    float*     gsq    = (float*)carve(512 * 4);
    float*     hstate = (float*)carve((size_t)BC * 128 * 4);
    float*     ghb    = (float*)carve((size_t)BC * 384 * 4);
    float*     pscore = (float*)carve((size_t)BC * 4);
    float*     pooled = (float*)carve((size_t)Bb * 128 * 4);
    _Float16*  Af16   = (_Float16*)carve((size_t)Nn * 512 * 2);  // A stage (f16)
    _Float16*  Btf16  = (_Float16*)carve((size_t)512 * 512 * 2); // B stage (f16, Nc x K)

    auto cvt = [&](const float* s, _Float16* d, size_t n) {
        k_cvt_f16<<<(int)((n + 255) / 256), 256, 0, stream>>>(s, d, n);
    };
    auto cvtT = [&](const float* s, _Float16* d, int R, int C) {
        size_t n = (size_t)R * C;
        k_cvt_f16_T<<<(int)((n + 255) / 256), 256, 0, stream>>>(s, d, R, C);
    };
    auto gemm = [&](const _Float16* A, int lda, const _Float16* Bt, int ldb,
                    float* Cc, int ldc, int M, int Ncol, int K, const float* bias) {
        dim3 g((Ncol + BN - 1) / BN, (M + BM - 1) / BM);
        k_gemm_wmma<<<g, 256, 0, stream>>>(A, lda, Bt, ldb, Cc, ldc, M, Ncol, K, bias);
    };

    auto gatLayer = [&](const float* feat, float* out, const float* a_s, const float* a_d,
                        const float* bias, int H, int F) {
        int waves  = Nn * H;
        int blocks = (waves + 7) / 8;
        k_node_scores<<<blocks, 256, 0, stream>>>(feat, a_s, a_d, ssrc, sdst, Nn, H, F);
        size_t nmh = (size_t)Nn * H;
        k_fill_u32<<<(int)((nmh + 255) / 256), 256, 0, stream>>>(mkey, 0x007FFFFFu, nmh); // key(-inf)
        k_fill_f32<<<(int)((nmh + 255) / 256), 256, 0, stream>>>(ssum, 0.f, nmh);
        size_t tot = (size_t)Nn * H * F;
        k_fill_bias<<<(int)((tot + 255) / 256), 256, 0, stream>>>(out, bias, tot, H * F);
        int eb = (ETOT + 255) / 256;
        k_edge_max<<<eb, 256, 0, stream>>>(ssrc, sdst, ei, E, ETOT, H, mkey);
        k_edge_expsum<<<eb, 256, 0, stream>>>(ssrc, sdst, ei, E, ETOT, H, mkey, ssum, alphaW);
        size_t tthreads = (size_t)ETOT * 32;
        k_edge_aggregate<<<(int)((tthreads + 255) / 256), 256, 0, stream>>>(
            feat, alphaW, ssum, ei, E, ETOT, H, F, out);
    };

    auto bnElu = [&](float* X, const float* g, const float* be, int C) {
        k_fill_f32<<<(C + 255) / 256, 256, 0, stream>>>(gsum, 0.f, (size_t)C);
        k_fill_f32<<<(C + 255) / 256, 256, 0, stream>>>(gsq, 0.f, (size_t)C);
        const int rpb = 256;
        k_bn_partial<<<(Nn + rpb - 1) / rpb, C, 0, stream>>>(X, Nn, C, gsum, gsq, rpb);
        k_bn_stats<<<1, C, 0, stream>>>(gsum, gsq, Nn, C);
        size_t tot = (size_t)Nn * C;
        k_bn_elu<<<(int)((tot + 255) / 256), 256, 0, stream>>>(X, gsum, gsq, g, be, tot, C);
    };

    // ---- GAT layer 1: 5 -> 4x128 concat ----
    cvt(x, Af16, (size_t)Nn * INF);
    cvtT(W1, Btf16, INF, HF);                      // Bt: 512 x 5
    gemm(Af16, INF, Btf16, INF, P, HF, Nn, HF, INF, nullptr);
    gatLayer(P, Q, as1, ad1, b1, 4, 128);
    bnElu(Q, g1, be1, HF);

    // ---- GAT layer 2: 512 -> 4x128 concat ----
    cvt(Q, Af16, (size_t)Nn * HF);
    cvtT(W2, Btf16, HF, HF);                       // Bt: 512 x 512
    gemm(Af16, HF, Btf16, HF, P, HF, Nn, HF, HF, nullptr);
    gatLayer(P, Q, as2, ad2, b2, 4, 128);
    bnElu(Q, g2, be2, HF);

    // ---- GAT layer 3: 512 -> 1x128 ----
    cvt(Q, Af16, (size_t)Nn * HF);
    cvtT(W3, Btf16, HF, 128);                      // Bt: 128 x 512
    gemm(Af16, HF, Btf16, HF, P, 128, Nn, 128, HF, nullptr);
    gatLayer(P, Q, as3, ad3, b3, 1, 128);
    bnElu(Q, g3, be3, 128);                        // Q[:Nn*128] == h3

    // ---- GRU: gi for all timesteps in one GEMM (A @ Wih^T) ----
    cvt(Q, Af16, (size_t)Nn * 128);
    cvt(Wih, Btf16, (size_t)384 * 128);            // already Nc x K layout
    gemm(Af16, 128, Btf16, 128, P, 384, Nn, 384, 128, nullptr);  // P == gi_all
    {
        size_t tot = (size_t)BC * 128;
        k_fill_f32<<<(int)((tot + 255) / 256), 256, 0, stream>>>(hstate, 0.f, tot);
    }
    cvt(Whh, Btf16, (size_t)384 * 128);            // reuse stage (stream-ordered)
    for (int t = 0; t < Tt; ++t) {
        cvt(hstate, Af16, (size_t)BC * 128);
        gemm(Af16, 128, Btf16, 128, ghb, 384, BC, 384, 128, nullptr); // h @ Whh^T
        int tot = BC * 128;
        k_gru_gates<<<(tot + 255) / 256, 256, 0, stream>>>(P, ghb, bih, bhh,
                                                           hstate, BC, Cch, t, Tt);
    }

    // ---- channel-attention pooling + final projection ----
    k_pool_score<<<BC, 64, 0, stream>>>(hstate, Wp1, bp1, Wp2, bp2, pscore);
    k_pool_apply<<<Bb, 128, 0, stream>>>(pscore, hstate, Cch, pooled);
    k_final<<<Bb, 64, 0, stream>>>(pooled, We, bemb, (float*)d_out);
}
